// SAGEPL_20100446946148
// MI455X (gfx1250) — matrix operand order repaired and checked by reading, not compiled, not run on hardware
//
#include <hip/hip_runtime.h>
#include <hip/hip_bf16.h>
#include <math.h>

#define N_NODES 50000
#define N_EDGES 800000
#define D_F     128          // D_IN == D_H == 128
#define D_OUT   47
#define NOISE_RATE 0.1f
#define NORM_EPS   1e-12f

typedef __bf16 bf16x16 __attribute__((ext_vector_type(16)));
typedef float  f32x8   __attribute__((ext_vector_type(8)));

// ---------------------------------------------------------------------------
// utility kernels
// ---------------------------------------------------------------------------
__global__ void zero_kernel(float* __restrict__ p, long n4) {
    long i = (long)blockIdx.x * blockDim.x + threadIdx.x;
    if (i < n4) {
        float4 z = make_float4(0.f, 0.f, 0.f, 0.f);
        reinterpret_cast<float4*>(p)[i] = z;
    }
}

__global__ void degree_kernel(const int* __restrict__ dst, float* __restrict__ cnt, int E) {
    int e = blockIdx.x * blockDim.x + threadIdx.x;
    if (e < E) unsafeAtomicAdd(&cnt[dst[e]], 1.0f);
}

// x_n = x + sign(x) * (noise / max(||noise||, eps)) * rate  -- one wave per node
__global__ void noise_kernel(const float* __restrict__ x, const float* __restrict__ noise,
                             float* __restrict__ xn, int N) {
    int t = blockIdx.x * blockDim.x + threadIdx.x;
    int node = t >> 5, lane = t & 31;
    if (node >= N) return;
    const float4 n4 = reinterpret_cast<const float4*>(noise + (size_t)node * D_F)[lane];
    float ss = n4.x*n4.x + n4.y*n4.y + n4.z*n4.z + n4.w*n4.w;
    #pragma unroll
    for (int off = 16; off > 0; off >>= 1) ss += __shfl_xor(ss, off, 32);
    float scl = NOISE_RATE / fmaxf(sqrtf(ss), NORM_EPS);
    const float4 x4 = reinterpret_cast<const float4*>(x + (size_t)node * D_F)[lane];
    float4 o4;
    #define SGN(v) ((v) > 0.f ? 1.f : ((v) < 0.f ? -1.f : 0.f))
    o4.x = x4.x + SGN(x4.x) * n4.x * scl;
    o4.y = x4.y + SGN(x4.y) * n4.y * scl;
    o4.z = x4.z + SGN(x4.z) * n4.z * scl;
    o4.w = x4.w + SGN(x4.w) * n4.w * scl;
    #undef SGN
    reinterpret_cast<float4*>(xn + (size_t)node * D_F)[lane] = o4;
}

// one wave per edge: 32 lanes x float4 = full 128-float row; L2-resident atomics
__global__ void scatter_kernel(const float* __restrict__ X, const int* __restrict__ src,
                               const int* __restrict__ dst, float* __restrict__ agg, int E) {
    long t = (long)blockIdx.x * blockDim.x + threadIdx.x;
    int e = (int)(t >> 5), lane = (int)(t & 31);
    if (e >= E) return;
    int s = src[e], d = dst[e];
    const float4 v = reinterpret_cast<const float4*>(X + (size_t)s * D_F)[lane];
    float* o = agg + (size_t)d * D_F + lane * 4;
    unsafeAtomicAdd(o + 0, v.x);
    unsafeAtomicAdd(o + 1, v.y);
    unsafeAtomicAdd(o + 2, v.z);
    unsafeAtomicAdd(o + 3, v.w);
}

// agg /= max(deg, 1)
__global__ void divcnt_kernel(float* __restrict__ agg, const float* __restrict__ cnt, long n4) {
    long i = (long)blockIdx.x * blockDim.x + threadIdx.x;
    if (i >= n4) return;
    long row = (i * 4) >> 7;                       // 128 floats per row
    float inv = 1.0f / fmaxf(cnt[row], 1.0f);
    float4 v = reinterpret_cast<float4*>(agg)[i];
    v.x *= inv; v.y *= inv; v.z *= inv; v.w *= inv;
    reinterpret_cast<float4*>(agg)[i] = v;
}

// ---------------------------------------------------------------------------
// WMMA GEMM: C[N,OUT] = Aagg @ Wl^T + Ax @ Wr^T + bias  (optional ReLU)
// K = 128. One wave per 16x16 output tile. bf16 A/B, f32 accumulate.
// ---------------------------------------------------------------------------
__global__ void sage_gemm_kernel(const float* __restrict__ Aagg, const float* __restrict__ Ax,
                                 const float* __restrict__ Wl,  const float* __restrict__ Wr,
                                 const float* __restrict__ bias, float* __restrict__ C,
                                 int rowTiles, int colTiles, int OUT, int doRelu) {
    int wid  = blockIdx.x * (blockDim.x >> 5) + (threadIdx.x >> 5);
    int lane = threadIdx.x & 31;
    if (wid >= rowTiles * colTiles) return;        // wave-uniform exit; EXEC full otherwise
    int rowTile = wid / colTiles;
    int colTile = wid - rowTile * colTiles;
    int nodeBase = rowTile << 4;
    int colBase  = colTile << 4;
    int halfsel  = lane >> 4;                      // 0: lanes 0-15, 1: lanes 16-31
    int rc       = lane & 15;                      // A row / B,C column within tile
    int o        = colBase + rc;                   // output feature (clamped for B reads)
    if (o > OUT - 1) o = OUT - 1;

    const float* ag = Aagg + (size_t)(nodeBase + rc) * D_F;
    const float* ax = Ax   + (size_t)(nodeBase + rc) * D_F;
    const float* wl = Wl   + (size_t)o * D_F + halfsel * 16;
    const float* wr = Wr   + (size_t)o * D_F + halfsel * 16;

    f32x8 acc = {};
    #pragma unroll
    for (int kb = 0; kb < D_F; kb += 32) {
        bf16x16 bl, br, aa, av;
        #pragma unroll
        for (int e = 0; e < 16; ++e) {             // B 32x16: lane holds col o, K=16*halfsel+e
            bl[e] = (__bf16)wl[kb + e];
            br[e] = (__bf16)wr[kb + e];
        }
        #pragma unroll
        for (int e = 0; e < 8; ++e) {              // A 16x32: K = 16*halfsel + e (contiguous)
            aa[e]     = (__bf16)ag[kb + halfsel * 16 + e];
            aa[8 + e] = (__bf16)ag[kb + halfsel * 16 + 8 + e];
            av[e]     = (__bf16)ax[kb + halfsel * 16 + e];
            av[8 + e] = (__bf16)ax[kb + halfsel * 16 + 8 + e];
        }
        acc = __builtin_amdgcn_wmma_f32_16x16x32_bf16(false, aa, false, bl, (short)0, acc, false, false);
        acc = __builtin_amdgcn_wmma_f32_16x16x32_bf16(false, av, false, br, (short)0, acc, false, false);
    }

    int col = colBase + rc;
    if (col < OUT) {                               // divergence only after all WMMAs
        float b = bias[col];
        #pragma unroll
        for (int r = 0; r < 8; ++r) {
            int m = r + halfsel * 8;               // C layout: VGPR r -> rows r / r+8
            float v = acc[r] + b;
            if (doRelu) v = fmaxf(v, 0.f);
            C[(size_t)(nodeBase + m) * OUT + col] = v;
        }
    }
}

// log_softmax over 47 classes -- one wave per node
__global__ void logsoftmax_kernel(const float* __restrict__ z, float* __restrict__ y, int N) {
    int t = blockIdx.x * blockDim.x + threadIdx.x;
    int node = t >> 5, lane = t & 31;
    if (node >= N) return;
    const float* zr = z + (size_t)node * D_OUT;
    float v0 = (lane      < D_OUT) ? zr[lane]      : -__builtin_inff();
    float v1 = (lane + 32 < D_OUT) ? zr[lane + 32] : -__builtin_inff();
    float m = fmaxf(v0, v1);
    #pragma unroll
    for (int off = 16; off > 0; off >>= 1) m = fmaxf(m, __shfl_xor(m, off, 32));
    float s = 0.f;
    if (lane      < D_OUT) s += __expf(v0 - m);
    if (lane + 32 < D_OUT) s += __expf(v1 - m);
    #pragma unroll
    for (int off = 16; off > 0; off >>= 1) s += __shfl_xor(s, off, 32);
    float lse = m + __logf(s);
    float* yr = y + (size_t)node * D_OUT;
    if (lane      < D_OUT) yr[lane]      = v0 - lse;
    if (lane + 32 < D_OUT) yr[lane + 32] = v1 - lse;
}

// ---------------------------------------------------------------------------
// launch
// ---------------------------------------------------------------------------
static inline int cdiv(long a, long b) { return (int)((a + b - 1) / b); }

extern "C" void kernel_launch(void* const* d_in, const int* in_sizes, int n_in,
                              void* d_out, int out_size, void* d_ws, size_t ws_size,
                              hipStream_t stream) {
    const float* x     = (const float*)d_in[0];
    const int*   eidx  = (const int*)  d_in[1];
    const float* noise = (const float*)d_in[2];
    const float* Wl0 = (const float*)d_in[3];
    const float* bl0 = (const float*)d_in[4];
    const float* Wr0 = (const float*)d_in[5];
    const float* Wl1 = (const float*)d_in[6];
    const float* bl1 = (const float*)d_in[7];
    const float* Wr1 = (const float*)d_in[8];
    const float* Wl2 = (const float*)d_in[9];
    const float* bl2 = (const float*)d_in[10];
    const float* Wr2 = (const float*)d_in[11];
    const int* src = eidx;              // edge_index[0]
    const int* dst = eidx + N_EDGES;    // edge_index[1]

    // outputs: (h_p, y_p, z_p, h_n, y_n, z_n) concatenated
    float* out = (float*)d_out;
    float* h_p = out;
    float* y_p = h_p + (size_t)N_NODES * D_F;
    float* z_p = y_p + (size_t)N_NODES * D_OUT;
    float* h_n = z_p + (size_t)N_NODES * D_OUT;
    float* y_n = h_n + (size_t)N_NODES * D_F;
    float* z_n = y_n + (size_t)N_NODES * D_OUT;

    // workspace layout
    float* ws  = (float*)d_ws;
    float* agg = ws;                               // N*128
    float* hid = agg + (size_t)N_NODES * D_F;      // N*128
    float* xnb = hid + (size_t)N_NODES * D_F;      // N*128
    float* cnt = xnb + (size_t)N_NODES * D_F;      // N

    const long nfeat  = (long)N_NODES * D_F;       // 6.4M floats
    const int  BLK    = 256;

    // degree (shared by both branches)
    zero_kernel<<<cdiv(N_NODES / 4, BLK), BLK, 0, stream>>>(cnt, N_NODES / 4);
    degree_kernel<<<cdiv(N_EDGES, BLK), BLK, 0, stream>>>(dst, cnt, N_EDGES);
    // noised input
    noise_kernel<<<cdiv((long)N_NODES * 32, BLK), BLK, 0, stream>>>(x, noise, xnb, N_NODES);

    const int rowTiles = N_NODES / 16;             // 3125 exact
    const int ctH = D_F / 16;                      // 8 col tiles (hidden)
    const int ctO = (D_OUT + 15) / 16;             // 3 col tiles (output)
    const int gemmBlkH = cdiv((long)rowTiles * ctH, 4);   // 4 waves / 128-thread block
    const int gemmBlkO = cdiv((long)rowTiles * ctO, 4);
    const int scatBlk  = cdiv((long)N_EDGES * 32, BLK);
    const int elemBlk  = cdiv(nfeat / 4, BLK);

    auto aggregate = [&](const float* X) {
        zero_kernel   <<<elemBlk, BLK, 0, stream>>>(agg, nfeat / 4);
        scatter_kernel<<<scatBlk, BLK, 0, stream>>>(X, src, dst, agg, N_EDGES);
        divcnt_kernel <<<elemBlk, BLK, 0, stream>>>(agg, cnt, nfeat / 4);
    };

    auto branch = [&](const float* X, float* hOut, float* yOut, float* zOut) {
        // layer 0: X -> hid (ReLU)
        aggregate(X);
        sage_gemm_kernel<<<gemmBlkH, 128, 0, stream>>>(agg, X, Wl0, Wr0, bl0, hid,
                                                       rowTiles, ctH, D_F, 1);
        // layer 1: hid -> hOut (ReLU)  [== "h" output]
        aggregate(hid);
        sage_gemm_kernel<<<gemmBlkH, 128, 0, stream>>>(agg, hid, Wl1, Wr1, bl1, hOut,
                                                       rowTiles, ctH, D_F, 1);
        // layer 2: hOut -> zOut (47 classes, no ReLU)
        aggregate(hOut);
        sage_gemm_kernel<<<gemmBlkO, 128, 0, stream>>>(agg, hOut, Wl2, Wr2, bl2, zOut,
                                                       rowTiles, ctO, D_OUT, 0);
        logsoftmax_kernel<<<cdiv((long)N_NODES * 32, BLK), BLK, 0, stream>>>(zOut, yOut, N_NODES);
    };

    branch(x,   h_p, y_p, z_p);   // clean branch
    branch(xnb, h_n, y_n, z_n);   // noised branch
}
